// AttentionLayer_91225105367774
// MI455X (gfx1250) — matrix operand order repaired and checked by reading, not compiled, not run on hardware
//
#include <hip/hip_runtime.h>
#include <hip/hip_bf16.h>

// ---------------------------------------------------------------------------
// GAT attention layer, B=8, N=2048, C=512, fp32 in/out.
//   s1 = x @ (w@a1), s2 = x @ (w@a2)        (collapses the x@w GEMM: 8.6 GF -> 34 MF)
//   m_i = lrelu(s1_i + max_j s2_j)          (per-batch max -> stable softmax, no rescale)
//   out[b,i,:] = (1/den_i) * sum_j exp(lrelu(s1_i+s2_j)-m_i) * x[b,j,:]
// Last line is flash-style with v_wmma_f32_16x16x32_bf16, double-buffered LDS.
// ---------------------------------------------------------------------------

typedef __attribute__((ext_vector_type(16))) __bf16 v16bf;
typedef __attribute__((ext_vector_type(8)))  float  v8f;

#define BDIM 8
#define NDIM 2048
#define CDIM 512

// workspace layout (float offsets)
#define WS_U1     0
#define WS_U2     512
#define WS_S1     1024
#define WS_S2     (1024 + 16384)
#define WS_RD     (1024 + 2*16384)
#define WS_S2MAX  (1024 + 3*16384)
#define WS_FLOATS 50192                 // padded to 16B multiple
// bf16 copy of x lives after the float region (16 MB)

#define XT_STRIDE 80                    // 32 bf16 (64B) + 16B pad -> bank spread
#define WT_STRIDE 80
#define XT_BUF    (CDIM * XT_STRIDE)    // 40960 B per buffer
#define WT_BUF    (32 * WT_STRIDE)      // 2560 B per buffer
#define SMEM_OFF_WT   (2 * XT_BUF)              // 81920
#define SMEM_OFF_S1L  (SMEM_OFF_WT + 2*WT_BUF)  // 87040
#define SMEM_OFF_ML   (SMEM_OFF_S1L + 128)      // 87168
#define SMEM_OFF_RDL  (SMEM_OFF_ML + 128)       // 87296
#define SMEM_BYTES    (SMEM_OFF_RDL + 128)      // 87424

__device__ __forceinline__ unsigned f2bf(float f) {          // RNE f32->bf16
    unsigned u = __float_as_uint(f);
    return (u + 0x7FFFu + ((u >> 16) & 1u)) >> 16;
}
__device__ __forceinline__ float lrelu(float v) { return v >= 0.f ? v : 0.2f * v; }

// ---- kernel 0: x (fp32) -> xbf (bf16), 8 elems/thread ----------------------
__global__ __launch_bounds__(256) void k_cvt(const float* __restrict__ x,
                                             unsigned short* __restrict__ xbf) {
    size_t idx = ((size_t)blockIdx.x * 256 + threadIdx.x) * 8;
    float4 f0 = *(const float4*)(x + idx);
    float4 f1 = *(const float4*)(x + idx + 4);
    uint4 o;
    o.x = f2bf(f0.x) | (f2bf(f0.y) << 16);
    o.y = f2bf(f0.z) | (f2bf(f0.w) << 16);
    o.z = f2bf(f1.x) | (f2bf(f1.y) << 16);
    o.w = f2bf(f1.z) | (f2bf(f1.w) << 16);
    *(uint4*)(xbf + idx) = o;
}

// ---- kernel 1: u1 = w@a1, u2 = w@a2 (C x C @ C) ----------------------------
__global__ __launch_bounds__(256) void k_u(const float* __restrict__ w,
                                           const float* __restrict__ a,
                                           float* __restrict__ u1,
                                           float* __restrict__ u2) {
    int i = blockIdx.x * 256 + threadIdx.x;
    if (i >= CDIM) return;
    float acc1 = 0.f, acc2 = 0.f;
    for (int c = 0; c < CDIM; ++c) {
        float wv = w[i * CDIM + c];
        acc1 += wv * a[c];
        acc2 += wv * a[CDIM + c];
    }
    u1[i] = acc1; u2[i] = acc2;
}

// ---- kernel 2: s1/s2 row dots, one wave32 per (b,n) row --------------------
__global__ __launch_bounds__(256) void k_s(const float* __restrict__ x,
                                           const float* __restrict__ u1,
                                           const float* __restrict__ u2,
                                           float* __restrict__ s1,
                                           float* __restrict__ s2) {
    int row  = blockIdx.x * 8 + (threadIdx.x >> 5);
    int lane = threadIdx.x & 31;
    const float* xr = x + (size_t)row * CDIM;
    float a1 = 0.f, a2 = 0.f;
    #pragma unroll
    for (int k = 0; k < 16; ++k) {
        int c = k * 32 + lane;
        float v = xr[c];
        a1 += v * u1[c];
        a2 += v * u2[c];
    }
    for (int off = 16; off; off >>= 1) {
        a1 += __shfl_xor(a1, off, 32);
        a2 += __shfl_xor(a2, off, 32);
    }
    if (!lane) { s1[row] = a1; s2[row] = a2; }
}

// ---- kernel 3: per-batch max of s2 -----------------------------------------
__global__ __launch_bounds__(256) void k_s2max(const float* __restrict__ s2,
                                               float* __restrict__ s2max) {
    int b = blockIdx.x, tid = threadIdx.x;
    __shared__ float red[256];
    float mx = -1e30f;
    for (int j = tid; j < NDIM; j += 256) mx = fmaxf(mx, s2[b * NDIM + j]);
    red[tid] = mx; __syncthreads();
    for (int s = 128; s > 0; s >>= 1) {
        if (tid < s) red[tid] = fmaxf(red[tid], red[tid + s]);
        __syncthreads();
    }
    if (!tid) s2max[b] = red[0];
}

// ---- kernel 4: rdenom[b,i] = 1 / sum_j exp(lrelu(s1+s2)-m), wave per row ---
__global__ __launch_bounds__(256) void k_rdenom(const float* __restrict__ s1,
                                                const float* __restrict__ s2,
                                                const float* __restrict__ s2max,
                                                float* __restrict__ rd) {
    int row  = blockIdx.x * 8 + (threadIdx.x >> 5);
    int lane = threadIdx.x & 31;
    int b    = row >> 11;                      // row / 2048
    float sv = s1[row];
    float mm = lrelu(sv + s2max[b]);
    const float* s2b = s2 + b * NDIM;
    float acc = 0.f;
    for (int j = lane; j < NDIM; j += 32)
        acc += __expf(lrelu(sv + s2b[j]) - mm);
    for (int off = 16; off; off >>= 1) acc += __shfl_xor(acc, off, 32);
    if (!lane) rd[row] = 1.0f / acc;
}

// ---- kernel 5: out = softmax(e) @ x  via WMMA bf16 -------------------------
// block = 256 thr (8 waves). Block tile: 32 rows x 512 cols of one batch.
// Wave owns 64 cols (4 c-tiles) x 2 i-tiles = 8 v8f accumulators.
// Double-buffered LDS: one barrier per 32-wide j-step; tile production for
// step k+1 overlaps the WMMA chain of step k.
__global__ __launch_bounds__(256) void k_attnx(const unsigned short* __restrict__ xbf,
                                               const float* __restrict__ s1,
                                               const float* __restrict__ s2,
                                               const float* __restrict__ s2max,
                                               const float* __restrict__ rdenom,
                                               float* __restrict__ out) {
    const int b    = blockIdx.y;
    const int i0   = blockIdx.x * 32;
    const int tid  = threadIdx.x;
    const int lane = tid & 31;
    const int wave = tid >> 5;
    const int m    = lane & 15;                // matrix row (A) / col (B,D)
    const int hi   = lane >> 4;                // lane half selects K/M block

    extern __shared__ unsigned char smem[];
    unsigned char* xt = smem;                          // xt[buf][c][j] bf16
    unsigned char* wt = smem + SMEM_OFF_WT;            // wt[buf][i][j] bf16
    float* s1l = (float*)(smem + SMEM_OFF_S1L);
    float* ml  = (float*)(smem + SMEM_OFF_ML);
    float* rdl = (float*)(smem + SMEM_OFF_RDL);

    if (tid < 32) {
        float v = s1[b * NDIM + i0 + tid];
        s1l[tid] = v;
        ml[tid]  = lrelu(v + s2max[b]);
        rdl[tid] = rdenom[b * NDIM + i0 + tid];
    }
    __syncthreads();

    // produce tiles for j-step j0 into buffer `buf`
    auto produce = [&](int j0, int buf) {
        // weight tile: 32x32, 4 elems/thread
        {
            int ii  = tid >> 3;
            int jj0 = (tid & 7) * 4;
            float sv = s1l[ii], mi = ml[ii];
            const float* s2p = s2 + b * NDIM + j0 + jj0;
            float w0 = __expf(lrelu(sv + s2p[0]) - mi);
            float w1 = __expf(lrelu(sv + s2p[1]) - mi);
            float w2 = __expf(lrelu(sv + s2p[2]) - mi);
            float w3 = __expf(lrelu(sv + s2p[3]) - mi);
            unsigned pk0 = f2bf(w0) | (f2bf(w1) << 16);
            unsigned pk1 = f2bf(w2) | (f2bf(w3) << 16);
            *(uint2*)(wt + buf * WT_BUF + ii * WT_STRIDE + jj0 * 2) =
                make_uint2(pk0, pk1);
        }
        // x tile transposed: 32 j x 512 c bf16 -> xt[c][j]
        {
            const unsigned short* xrow = xbf + (size_t)(b * NDIM + j0) * CDIM;
            unsigned char* xb = xt + buf * XT_BUF;
            #pragma unroll
            for (int r = 0; r < 8; ++r) {
                int T  = tid + 256 * r;
                int c4 = T & 127;             // c = c4*4 .. +3
                int jp = T >> 7;              // j = 2*jp, 2*jp+1
                const unsigned short* p0 = xrow + (2 * jp) * CDIM + c4 * 4;
                uint2 A  = *(const uint2*)p0;
                uint2 Bv = *(const uint2*)(p0 + CDIM);
                unsigned o0 = (A.x & 0xffffu) | (Bv.x << 16);
                unsigned o1 = (A.x >> 16)     | (Bv.x & 0xffff0000u);
                unsigned o2 = (A.y & 0xffffu) | (Bv.y << 16);
                unsigned o3 = (A.y >> 16)     | (Bv.y & 0xffff0000u);
                unsigned char* q = xb + (c4 * 4) * XT_STRIDE + jp * 4;
                *(unsigned*)(q)                 = o0;
                *(unsigned*)(q + XT_STRIDE)     = o1;
                *(unsigned*)(q + 2 * XT_STRIDE) = o2;
                *(unsigned*)(q + 3 * XT_STRIDE) = o3;
            }
        }
    };

    v8f acc[2][4];
    #pragma unroll
    for (int it = 0; it < 2; ++it)
        #pragma unroll
        for (int ct = 0; ct < 4; ++ct) acc[it][ct] = {};

    const int c0 = wave * 64;

    produce(0, 0);

    #pragma unroll 1
    for (int k = 0; k < NDIM / 32; ++k) {
        const int buf = k & 1;
        __syncthreads();    // tiles[buf] ready; tiles[buf^1] consumption done

        // -- hoist ALL fragments (distinct regs -> loads overlap, WMMAs b2b) --
        union Frag { uint4 q[2]; v16bf v; };
        Frag afr[2], bfr[4];
        const unsigned char* wbase = wt + buf * WT_BUF;
        const unsigned char* xbase = xt + buf * XT_BUF;
        #pragma unroll
        for (int it = 0; it < 2; ++it) {
            // A 16x32: lane half hi -> K = 16*hi .. 16*hi+15 contiguous
            const unsigned char* rp = wbase + (m + 16 * it) * WT_STRIDE + hi * 32;
            afr[it].q[0] = *(const uint4*)(rp);
            afr[it].q[1] = *(const uint4*)(rp + 16);
        }
        #pragma unroll
        for (int ct = 0; ct < 4; ++ct) {
            // B 32x16: lane = column, K = 16*hi .. 16*hi+15 contiguous
            const unsigned char* rp = xbase + (c0 + ct * 16 + m) * XT_STRIDE + hi * 32;
            bfr[ct].q[0] = *(const uint4*)(rp);
            bfr[ct].q[1] = *(const uint4*)(rp + 16);
        }

        // -- overlap: produce next tile while this step's WMMAs run ----------
        if (k + 1 < NDIM / 32) produce((k + 1) * 32, buf ^ 1);
        // prefetch tile k+2 toward this WGP (global_prefetch_b8); x sits in L2
        if (k + 2 < NDIM / 32) {
            const unsigned char* nx = (const unsigned char*)
                (xbf + (size_t)(b * NDIM + (k + 2) * 32) * CDIM);
            __builtin_prefetch(nx + tid * 128, 0, 1);
        }

        // -- 8 back-to-back WMMAs --------------------------------------------
        #pragma unroll
        for (int ct = 0; ct < 4; ++ct)
            #pragma unroll
            for (int it = 0; it < 2; ++it)
                acc[it][ct] = __builtin_amdgcn_wmma_f32_16x16x32_bf16(
                    false, afr[it].v, false, bfr[ct].v,
                    (short)0, acc[it][ct], false, false);
    }

    // -- scale by 1/denominator, store (C/D layout: M = r + 8*hi, N = m) ------
    #pragma unroll
    for (int it = 0; it < 2; ++it) {
        #pragma unroll
        for (int r = 0; r < 8; ++r) {
            int row = it * 16 + r + 8 * hi;
            float sc = rdl[row];
            size_t base = ((size_t)(b * NDIM + i0 + row)) * CDIM;
            #pragma unroll
            for (int ct = 0; ct < 4; ++ct)
                out[base + c0 + ct * 16 + m] = acc[it][ct][r] * sc;
        }
    }
}

// ---------------------------------------------------------------------------
extern "C" void kernel_launch(void* const* d_in, const int* in_sizes, int n_in,
                              void* d_out, int out_size, void* d_ws, size_t ws_size,
                              hipStream_t stream) {
    const float* x = (const float*)d_in[0];   // (8,2048,512)
    const float* w = (const float*)d_in[1];   // (512,512)
    const float* a = (const float*)d_in[2];   // (1024,1)
    float* out = (float*)d_out;

    float* wsf = (float*)d_ws;
    float* u1     = wsf + WS_U1;
    float* u2     = wsf + WS_U2;
    float* s1     = wsf + WS_S1;
    float* s2     = wsf + WS_S2;
    float* rd     = wsf + WS_RD;
    float* s2max  = wsf + WS_S2MAX;
    unsigned short* xbf = (unsigned short*)((char*)d_ws + (size_t)WS_FLOATS * 4);

    // 0) fp32 -> bf16 copy of x (8M elems, 8 per thread)
    k_cvt<<<4096, 256, 0, stream>>>(x, xbf);
    // 1) u1 = w@a1, u2 = w@a2
    k_u<<<2, 256, 0, stream>>>(w, a, u1, u2);
    // 2) s1/s2 = x@u1, x@u2 (wave per row)
    k_s<<<2048, 256, 0, stream>>>(x, u1, u2, s1, s2);
    // 3) per-batch max of s2
    k_s2max<<<BDIM, 256, 0, stream>>>(s2, s2max);
    // 4) softmax reciprocal denominators
    k_rdenom<<<2048, 256, 0, stream>>>(s1, s2, s2max, rd);
    // 5) flash-style attention @ x with bf16 WMMA (double-buffered LDS)
    dim3 grid(NDIM / 32, BDIM);
    k_attnx<<<grid, 256, SMEM_BYTES, stream>>>(xbf, s1, s2, s2max, rd, out);
}
